// BidirectionalSelfFlashAttention_64939905516272
// MI455X (gfx1250) — matrix operand order repaired and checked by reading, not compiled, not run on hardware
//
#include <hip/hip_runtime.h>

#define B_ 4
#define T_ 2048
#define C_ 1024
#define H_ 16
#define D_ 64
#define M_ (B_*T_)
#define N3_ (3*C_)

typedef __attribute__((ext_vector_type(16))) __bf16 v16bf;
typedef __attribute__((ext_vector_type(8)))  float  v8f;
typedef __attribute__((ext_vector_type(4)))  unsigned int v4u;
typedef __attribute__((ext_vector_type(4)))  int v4i;

// address-space qualified 16B-vector pointers for the async builtin
typedef __attribute__((address_space(1))) v4i as1_v4i;
typedef __attribute__((address_space(3))) v4i as3_v4i;

union Frag16 { v16bf bf; v4u q[2]; };

__device__ __forceinline__ unsigned short f32_to_bf16(float f) {
  union { float f; unsigned u; } v; v.f = f;
  unsigned r = v.u + 0x7FFFu + ((v.u >> 16) & 1u);
  return (unsigned short)(r >> 16);
}

__device__ __forceinline__ v8f vzero8() {
  v8f z;
#pragma unroll
  for (int i = 0; i < 8; ++i) z[i] = 0.0f;
  return z;
}

__device__ __forceinline__ v8f wmma_bf16(Frag16 a, Frag16 b, v8f c) {
  return __builtin_amdgcn_wmma_f32_16x16x32_bf16(
      false, a.bf, false, b.bf, (short)0, c, false, false);
}

// ---- gfx1250 async global->LDS copy (16B per lane), with safe fallback ----
__device__ __forceinline__ void cp_async16(void* lds, const void* glb) {
#if __has_builtin(__builtin_amdgcn_global_load_async_to_lds_b128)
  __builtin_amdgcn_global_load_async_to_lds_b128(
      (as1_v4i*)glb, (as3_v4i*)lds, 0, 0);
#else
  *(v4u*)lds = *(const v4u*)glb;
#endif
}

__device__ __forceinline__ void async_wait0() {
#if __has_builtin(__builtin_amdgcn_global_load_async_to_lds_b128)
#if __has_builtin(__builtin_amdgcn_s_wait_asynccnt)
  __builtin_amdgcn_s_wait_asynccnt(0);
#else
  asm volatile("s_wait_asynccnt 0" ::: "memory");
#endif
#endif
}

// ---------------------------------------------------------------------------
// Kernel 0a: one-shot fp32 -> bf16 conversion (x).
// ---------------------------------------------------------------------------
__global__ __launch_bounds__(256) void cvt_bf16_kernel(
    const float* __restrict__ in, unsigned short* __restrict__ out) {
  int i = (blockIdx.x * 256 + threadIdx.x) * 4;
  float4 v = *(const float4*)(in + i);
  unsigned a0 = f32_to_bf16(v.x), a1 = f32_to_bf16(v.y);
  unsigned a2 = f32_to_bf16(v.z), a3 = f32_to_bf16(v.w);
  unsigned int* o = (unsigned int*)(out + i);
  o[0] = a0 | (a1 << 16);
  o[1] = a2 | (a3 << 16);
}

// ---------------------------------------------------------------------------
// Kernel 0b: one-shot fp32 [K][N] -> bf16 transposed [N][K] (weights).
// ---------------------------------------------------------------------------
__global__ __launch_bounds__(256) void cvtT_bf16_kernel(
    const float* __restrict__ in, unsigned short* __restrict__ out,
    int N, int K) {
  __shared__ unsigned short tile[64][65];
  const int tid = threadIdx.x;
  const int k0 = blockIdx.y * 64, n0 = blockIdx.x * 64;
#pragma unroll
  for (int e = 0; e < 16; ++e) {
    int idx = e * 256 + tid;
    int r = idx >> 6, c = idx & 63;  // r: k, c: n  (coalesced read along n)
    tile[c][r] = f32_to_bf16(in[(size_t)(k0 + r) * N + n0 + c]);
  }
  __syncthreads();
#pragma unroll
  for (int e = 0; e < 16; ++e) {
    int idx = e * 256 + tid;
    int r = idx >> 6, c = idx & 63;  // r: n, c: k  (coalesced write along k)
    out[(size_t)(n0 + r) * K + k0 + c] = tile[r][c];
  }
}

// ---------------------------------------------------------------------------
// Kernel 1: fused QKV projection.  xb(bf16)[M_,C_] x WqkvT(bf16)[3C_,C_]^T
// + bias -> Q/K bf16 [B,H,T,D], V bf16 [B,H,D,T] (pre-transposed for PV).
// 128x256 WG tile, BK=32, 8 waves (2x4), each wave a 64x64 subtile =
// 4x4 WMMA accumulators -> 16 ds_load_b128 per 16 WMMAs.
// ---------------------------------------------------------------------------
__global__ __launch_bounds__(256) void qkv_proj_kernel(
    const unsigned short* __restrict__ xb, const unsigned short* __restrict__ wT,
    const float* __restrict__ bias,
    unsigned short* __restrict__ Qo, unsigned short* __restrict__ Ko,
    unsigned short* __restrict__ Vo) {
  __shared__ __align__(16) unsigned short sA[128 * 32];  // [m][k]
  __shared__ __align__(16) unsigned short sB[256 * 32];  // [n][k]
  const int tid  = threadIdx.x;
  const int wave = tid >> 5, lane = tid & 31;
  const int half = lane >> 4, l16 = lane & 15;
  const int m0 = blockIdx.y * 128, n0 = blockIdx.x * 256;
  const int wm = wave & 1, wn = wave >> 1;  // wm: 0..1 (64 rows), wn: 0..3 (64 cols)

  v8f acc[4][4];
#pragma unroll
  for (int i = 0; i < 4; ++i)
#pragma unroll
    for (int j = 0; j < 4; ++j) acc[i][j] = vzero8();

  for (int k0 = 0; k0 < C_; k0 += 32) {
#pragma unroll
    for (int e = 0; e < 2; ++e) {  // A tile: 128x32 = 512x16B
      int idx = e * 256 + tid;
      int r = idx >> 2, c4 = (idx & 3) * 8;
      cp_async16(sA + r * 32 + c4, xb + (size_t)(m0 + r) * C_ + k0 + c4);
    }
#pragma unroll
    for (int e = 0; e < 4; ++e) {  // B tile: 256x32 = 1024x16B from [n][k]
      int idx = e * 256 + tid;
      int r = idx >> 2, c4 = (idx & 3) * 8;
      cp_async16(sB + r * 32 + c4, wT + (size_t)(n0 + r) * C_ + k0 + c4);
    }
    async_wait0();
    __syncthreads();

    Frag16 fa[4], fb[4];
#pragma unroll
    for (int i = 0; i < 4; ++i) {
      const unsigned short* p = sA + (wm * 64 + i * 16 + l16) * 32;
      fa[i].q[0] = *(const v4u*)(p + half * 8);
      fa[i].q[1] = *(const v4u*)(p + 16 + half * 8);
    }
#pragma unroll
    for (int j = 0; j < 4; ++j) {
      const unsigned short* p = sB + (wn * 64 + j * 16 + l16) * 32 + half * 8;
      fb[j].q[0] = *(const v4u*)(p);
      fb[j].q[1] = *(const v4u*)(p + 16);
    }
#pragma unroll
    for (int i = 0; i < 4; ++i)
#pragma unroll
      for (int j = 0; j < 4; ++j)
        acc[i][j] = wmma_bf16(fa[i], fb[j], acc[i][j]);
    __syncthreads();
  }

  // Epilogue: bias, bf16, scatter.  n-block (256 wide) never straddles the
  // Q/K/V boundaries (multiples of 1024), so `which` is block-uniform.
#pragma unroll
  for (int j = 0; j < 4; ++j) {
    int n = n0 + wn * 64 + j * 16 + l16;  // [0, 3072)
    float bv = bias[n];
    int which = n >> 10;
    int cc = n & 1023;
    int h = cc >> 6, d = cc & 63;
#pragma unroll
    for (int i = 0; i < 4; ++i)
#pragma unroll
      for (int r = 0; r < 8; ++r) {
        int m = m0 + wm * 64 + i * 16 + r + half * 8;
        int b = m >> 11, tt = m & 2047;
        unsigned short val = f32_to_bf16(acc[i][j][r] + bv);
        if (which == 0)
          Qo[(((size_t)(b * H_ + h)) * T_ + tt) * D_ + d] = val;
        else if (which == 1)
          Ko[(((size_t)(b * H_ + h)) * T_ + tt) * D_ + d] = val;
        else
          Vo[(((size_t)(b * H_ + h)) * D_ + d) * T_ + tt] = val;  // [B,H,D,T]
      }
  }
}

// ---------------------------------------------------------------------------
// Kernel 2: flash attention.  WG = 4 waves = 64 query rows of one (b,h).
// K block in LDS [s][d]; V already stored transposed -> straight block copy.
// ---------------------------------------------------------------------------
__global__ __launch_bounds__(128) void attn_kernel(
    const unsigned short* __restrict__ Qg, const unsigned short* __restrict__ Kg,
    const unsigned short* __restrict__ VTg, const float* __restrict__ maskp,
    unsigned short* __restrict__ Y) {
  __shared__ __align__(16) unsigned short sQ[64 * 64];
  __shared__ __align__(16) unsigned short sK[64 * 64];
  __shared__ __align__(16) unsigned short sVT[64 * 64];  // [d][s]
  __shared__ __align__(16) unsigned short sP[4][16 * 64];

  const int tid  = threadIdx.x;
  const int wave = tid >> 5, lane = tid & 31;
  const int half = lane >> 4, l16 = lane & 15;
  const int bh = blockIdx.y;
  const int b = bh >> 4, h = bh & 15;
  const int q0 = blockIdx.x * 64;
  const size_t headbase  = ((size_t)(b * H_ + h)) * T_ * D_;  // Q,K [T][D]
  const size_t headbaseV = ((size_t)(b * H_ + h)) * D_ * T_;  // VT  [D][T]

#pragma unroll
  for (int e = 0; e < 4; ++e) {  // Q tile 64x64
    int idx = e * 128 + tid;
    int r = idx >> 3, c = (idx & 7) * 8;
    cp_async16(sQ + r * 64 + c, Qg + headbase + (size_t)(q0 + r) * D_ + c);
  }
  async_wait0();
  __syncthreads();

  Frag16 aq[2];
  {
    const unsigned short* p = sQ + (wave * 16 + l16) * 64;
#pragma unroll
    for (int cx = 0; cx < 2; ++cx) {
      aq[cx].q[0] = *(const v4u*)(p + cx * 32 + half * 8);
      aq[cx].q[1] = *(const v4u*)(p + cx * 32 + 16 + half * 8);
    }
  }

  v8f rmax, rsum, O[4];
#pragma unroll
  for (int r = 0; r < 8; ++r) { rmax[r] = -3.0e30f; rsum[r] = 0.0f; }
#pragma unroll
  for (int nt = 0; nt < 4; ++nt) O[nt] = vzero8();

  const float scale = 0.125f;  // 1/sqrt(64)

  for (int s0 = 0; s0 < T_; s0 += 64) {
#pragma unroll
    for (int e = 0; e < 4; ++e) {  // K block [s][d]
      int idx = e * 128 + tid;
      int r = idx >> 3, c = (idx & 7) * 8;
      cp_async16(sK + r * 64 + c, Kg + headbase + (size_t)(s0 + r) * D_ + c);
    }
#pragma unroll
    for (int e = 0; e < 4; ++e) {  // V block already [d][s]
      int idx = e * 128 + tid;
      int r = idx >> 3, c = (idx & 7) * 8;
      cp_async16(sVT + r * 64 + c, VTg + headbaseV + (size_t)r * T_ + s0 + c);
    }
    async_wait0();
    __syncthreads();

    // ---- scores S = scale * Q K^T + mask ----
    v8f S[4];
#pragma unroll
    for (int nt = 0; nt < 4; ++nt) {
      const unsigned short* kr = sK + (nt * 16 + l16) * 64;
      Frag16 b0, b1;
      b0.q[0] = *(const v4u*)(kr + half * 8);
      b0.q[1] = *(const v4u*)(kr + 16 + half * 8);
      b1.q[0] = *(const v4u*)(kr + 32 + half * 8);
      b1.q[1] = *(const v4u*)(kr + 32 + 16 + half * 8);
      v8f z = vzero8();
      z = wmma_bf16(aq[0], b0, z);
      z = wmma_bf16(aq[1], b1, z);
      float mv = maskp[b * T_ + s0 + nt * 16 + l16];
#pragma unroll
      for (int r = 0; r < 8; ++r) z[r] = z[r] * scale + mv;
      S[nt] = z;
    }

    // ---- online softmax (row stats per VGPR element, xor-reduce in halves)
    v8f tmax;
#pragma unroll
    for (int r = 0; r < 8; ++r)
      tmax[r] = fmaxf(fmaxf(S[0][r], S[1][r]), fmaxf(S[2][r], S[3][r]));
#pragma unroll
    for (int r = 0; r < 8; ++r) {
      float v = tmax[r];
      v = fmaxf(v, __shfl_xor(v, 1));
      v = fmaxf(v, __shfl_xor(v, 2));
      v = fmaxf(v, __shfl_xor(v, 4));
      v = fmaxf(v, __shfl_xor(v, 8));
      tmax[r] = v;
    }
    v8f nmax, alpha;
#pragma unroll
    for (int r = 0; r < 8; ++r) {
      nmax[r] = fmaxf(rmax[r], tmax[r]);
      alpha[r] = __expf(rmax[r] - nmax[r]);
    }
    v8f tsum = vzero8();
#pragma unroll
    for (int nt = 0; nt < 4; ++nt)
#pragma unroll
      for (int r = 0; r < 8; ++r) {
        float p = __expf(S[nt][r] - nmax[r]);
        S[nt][r] = p;
        tsum[r] += p;
      }
#pragma unroll
    for (int r = 0; r < 8; ++r) {
      float v = tsum[r];
      v += __shfl_xor(v, 1);
      v += __shfl_xor(v, 2);
      v += __shfl_xor(v, 4);
      v += __shfl_xor(v, 8);
      tsum[r] = v;
    }
#pragma unroll
    for (int r = 0; r < 8; ++r) {
      rsum[r] = rsum[r] * alpha[r] + tsum[r];
      rmax[r] = nmax[r];
    }
#pragma unroll
    for (int nt = 0; nt < 4; ++nt)
#pragma unroll
      for (int r = 0; r < 8; ++r) O[nt][r] *= alpha[r];

    // ---- P (C-layout fp32) -> LDS bf16 -> A-fragments ----
    unsigned short* pw = sP[wave];
#pragma unroll
    for (int nt = 0; nt < 4; ++nt)
#pragma unroll
      for (int r = 0; r < 8; ++r)
        pw[(r + half * 8) * 64 + nt * 16 + l16] = f32_to_bf16(S[nt][r]);
    asm volatile("s_wait_dscnt 0" ::: "memory");  // per-wave LDS RAW

    Frag16 ap[2];
    {
      const unsigned short* p = pw + l16 * 64;
#pragma unroll
      for (int cx = 0; cx < 2; ++cx) {
        ap[cx].q[0] = *(const v4u*)(p + cx * 32 + half * 8);
        ap[cx].q[1] = *(const v4u*)(p + cx * 32 + 16 + half * 8);
      }
    }
    // ---- O += P V ----
#pragma unroll
    for (int nt = 0; nt < 4; ++nt) {
      const unsigned short* vr = sVT + (nt * 16 + l16) * 64;
      Frag16 v0, v1;
      v0.q[0] = *(const v4u*)(vr + half * 8);
      v0.q[1] = *(const v4u*)(vr + 16 + half * 8);
      v1.q[0] = *(const v4u*)(vr + 32 + half * 8);
      v1.q[1] = *(const v4u*)(vr + 32 + 16 + half * 8);
      O[nt] = wmma_bf16(ap[0], v0, O[nt]);
      O[nt] = wmma_bf16(ap[1], v1, O[nt]);
    }
    __syncthreads();
  }

  // normalize and write merged-head Y (bf16, [B*T, C])
#pragma unroll
  for (int nt = 0; nt < 4; ++nt) {
    int col = h * 64 + nt * 16 + l16;
#pragma unroll
    for (int r = 0; r < 8; ++r) {
      int row = q0 + wave * 16 + r + half * 8;
      float val = O[nt][r] / rsum[r];
      Y[((size_t)(b * T_ + row)) * C_ + col] = f32_to_bf16(val);
    }
  }
}

// ---------------------------------------------------------------------------
// Kernel 3: output projection.  Y(bf16)[M_,C_] x WoT(bf16)[C_,C_]^T + bo.
// Same 128x256 / 64x64-per-wave tiling as kernel 1.
// ---------------------------------------------------------------------------
__global__ __launch_bounds__(256) void out_proj_kernel(
    const unsigned short* __restrict__ Y, const unsigned short* __restrict__ wT,
    const float* __restrict__ bias, float* __restrict__ out) {
  __shared__ __align__(16) unsigned short sA[128 * 32];
  __shared__ __align__(16) unsigned short sB[256 * 32];
  const int tid  = threadIdx.x;
  const int wave = tid >> 5, lane = tid & 31;
  const int half = lane >> 4, l16 = lane & 15;
  const int m0 = blockIdx.y * 128, n0 = blockIdx.x * 256;
  const int wm = wave & 1, wn = wave >> 1;

  v8f acc[4][4];
#pragma unroll
  for (int i = 0; i < 4; ++i)
#pragma unroll
    for (int j = 0; j < 4; ++j) acc[i][j] = vzero8();

  for (int k0 = 0; k0 < C_; k0 += 32) {
#pragma unroll
    for (int e = 0; e < 2; ++e) {
      int idx = e * 256 + tid;
      int r = idx >> 2, c4 = (idx & 3) * 8;
      cp_async16(sA + r * 32 + c4, Y + (size_t)(m0 + r) * C_ + k0 + c4);
    }
#pragma unroll
    for (int e = 0; e < 4; ++e) {
      int idx = e * 256 + tid;
      int r = idx >> 2, c4 = (idx & 3) * 8;
      cp_async16(sB + r * 32 + c4, wT + (size_t)(n0 + r) * C_ + k0 + c4);
    }
    async_wait0();
    __syncthreads();

    Frag16 fa[4], fb[4];
#pragma unroll
    for (int i = 0; i < 4; ++i) {
      const unsigned short* p = sA + (wm * 64 + i * 16 + l16) * 32;
      fa[i].q[0] = *(const v4u*)(p + half * 8);
      fa[i].q[1] = *(const v4u*)(p + 16 + half * 8);
    }
#pragma unroll
    for (int j = 0; j < 4; ++j) {
      const unsigned short* p = sB + (wn * 64 + j * 16 + l16) * 32 + half * 8;
      fb[j].q[0] = *(const v4u*)(p);
      fb[j].q[1] = *(const v4u*)(p + 16);
    }
#pragma unroll
    for (int i = 0; i < 4; ++i)
#pragma unroll
      for (int j = 0; j < 4; ++j)
        acc[i][j] = wmma_bf16(fa[i], fb[j], acc[i][j]);
    __syncthreads();
  }

#pragma unroll
  for (int j = 0; j < 4; ++j) {
    int n = n0 + wn * 64 + j * 16 + l16;
    float bv = bias[n];
#pragma unroll
    for (int i = 0; i < 4; ++i)
#pragma unroll
      for (int r = 0; r < 8; ++r) {
        int m = m0 + wm * 64 + i * 16 + r + half * 8;
        out[(size_t)m * C_ + n] = acc[i][j][r] + bv;
      }
  }
}

// ---------------------------------------------------------------------------
extern "C" void kernel_launch(void* const* d_in, const int* in_sizes, int n_in,
                              void* d_out, int out_size, void* d_ws,
                              size_t ws_size, hipStream_t stream) {
  (void)in_sizes; (void)n_in; (void)out_size; (void)ws_size;
  const float* x    = (const float*)d_in[0];
  const float* mask = (const float*)d_in[1];
  const float* Wqkv = (const float*)d_in[2];
  const float* bqkv = (const float*)d_in[3];
  const float* Wo   = (const float*)d_in[4];
  const float* bo   = (const float*)d_in[5];
  float* out = (float*)d_out;

  // workspace (all bf16/ushort):
  //   xb[M_,C_] | WqkvT[3C_,C_] | WoT[C_,C_] | Q | K | VT | Y   (~88 MB)
  unsigned short* xb     = (unsigned short*)d_ws;
  unsigned short* WqkvT  = xb    + (size_t)M_ * C_;
  unsigned short* WoT    = WqkvT + (size_t)N3_ * C_;
  unsigned short* Qb     = WoT   + (size_t)C_ * C_;
  unsigned short* Kb     = Qb    + (size_t)M_ * C_;
  unsigned short* VTb    = Kb    + (size_t)M_ * C_;
  unsigned short* Yb     = VTb   + (size_t)M_ * C_;

  cvt_bf16_kernel<<<(M_ * C_) / 1024, 256, 0, stream>>>(x, xb);
  cvtT_bf16_kernel<<<dim3(N3_ / 64, C_ / 64), 256, 0, stream>>>(Wqkv, WqkvT,
                                                                N3_, C_);
  cvtT_bf16_kernel<<<dim3(C_ / 64, C_ / 64), 256, 0, stream>>>(Wo, WoT,
                                                               C_, C_);
  qkv_proj_kernel<<<dim3(N3_ / 256, M_ / 128), 256, 0, stream>>>(
      xb, WqkvT, bqkv, Qb, Kb, VTb);
  attn_kernel<<<dim3(T_ / 64, B_ * H_), 128, 0, stream>>>(
      Qb, Kb, VTb, mask, Yb);
  out_proj_kernel<<<dim3(C_ / 256, M_ / 128), 256, 0, stream>>>(
      Yb, WoT, bo, out);
}
